// GCN_56384330662316
// MI455X (gfx1250) — matrix operand order repaired and checked by reading, not compiled, not run on hardware
//
#include <hip/hip_runtime.h>
#include <hip/hip_bf16.h>

typedef __bf16 bf16;
typedef __attribute__((ext_vector_type(16))) __bf16 v16bf;
typedef __attribute__((ext_vector_type(8)))  __bf16 v8bf;
typedef __attribute__((ext_vector_type(4)))  __bf16 v4bf;
typedef __attribute__((ext_vector_type(8)))  float  v8f;
typedef __attribute__((ext_vector_type(4)))  int    v4i;

#define N_NODES 8192
#define C_DIM   512
#define BM 128
#define BN 128
#define BK 32

// CDNA5 async global->LDS path (guarded: falls back to VGPR staging if the
// toolchain lacks the builtins).
#if defined(__HIP_DEVICE_COMPILE__)
#if __has_builtin(__builtin_amdgcn_global_load_async_to_lds_b128) && \
    __has_builtin(__builtin_amdgcn_s_wait_asynccnt)
#define GCN_ASYNC_LDS 1
#endif
#endif

#if defined(GCN_ASYNC_LDS)
typedef __attribute__((address_space(1))) v4i g_v4i;   // global int4
typedef __attribute__((address_space(3))) v4i l_v4i;   // LDS int4
__device__ __forceinline__ void gcn_async_b128(const void* gsrc, void* ldst) {
    __builtin_amdgcn_global_load_async_to_lds_b128(
        (g_v4i*)gsrc, (l_v4i*)ldst, 0, 0);
}
#endif

// ---------------- workspace layout (bytes) ----------------
// [0,4)        : int has_zero flag
// [256, 33024) : inv_sr  (f32[8192])  = rsqrt(row_degree)
// [33024,65792): inv_sc  (f32[8192])  = rsqrt(col_degree)
// [65792, ...) : hs_t    (bf16[512][8192]) scaled transformed features
#define WS_OFF_INVSR 256
#define WS_OFF_INVSC 33024
#define WS_OFF_HST   65792

__global__ void gcn_init_flag(int* flag) { *flag = 0; }

// one block per row: row_degree[i] = sum_j adj[i][j]
__global__ __launch_bounds__(256)
void gcn_row_degree(const float* __restrict__ adj, float* __restrict__ inv_sr,
                    int* __restrict__ flag) {
    __shared__ float red[256];
    const int row = blockIdx.x;
    const float* p = adj + (size_t)row * N_NODES;
    float s = 0.f;
    for (int k = threadIdx.x; k < N_NODES; k += 256) s += p[k];
    red[threadIdx.x] = s;
    __syncthreads();
    for (int off = 128; off > 0; off >>= 1) {
        if (threadIdx.x < off) red[threadIdx.x] += red[threadIdx.x + off];
        __syncthreads();
    }
    if (threadIdx.x == 0) {
        float d = red[0];
        if (d == 0.f) atomicOr(flag, 1);
        inv_sr[row] = rsqrtf(d);
    }
}

// col_degree[j] = sum_i adj[i][j]; one column per thread, coalesced rows
__global__ __launch_bounds__(256)
void gcn_col_degree(const float* __restrict__ adj, float* __restrict__ inv_sc,
                    int* __restrict__ flag) {
    const int j = blockIdx.x * 256 + threadIdx.x;
    float s = 0.f;
    for (int i = 0; i < N_NODES; ++i) s += adj[(size_t)i * N_NODES + j];
    if (s == 0.f) atomicOr(flag, 1);
    inv_sc[j] = rsqrtf(s);
}

// GEMM-1: hs_t[c][j] = bf16( relu(feat[j,:] . W[c,:] + b[c]) * inv_sc[j] )
__global__ __launch_bounds__(256)
void gcn_feat_xform(const float* __restrict__ feat,
                    const float* __restrict__ W,
                    const float* __restrict__ bias,
                    const float* __restrict__ inv_sc,
                    bf16* __restrict__ hs_t) {
    __shared__ __align__(32) bf16 As[2][BM][BK];
    __shared__ __align__(32) bf16 Bs[2][BN][BK];
    const int tid  = threadIdx.x;
    const int wave = tid >> 5;
    const int lane = tid & 31;
    const int lh   = lane >> 4;   // K-half select
    const int lm   = lane & 15;   // row/col within 16
    const int m0   = blockIdx.y * BM;
    const int n0   = blockIdx.x * BN;

    v8f acc[8] = {};
    float4 aReg[4], bReg[4];

    // prologue: tile 0 -> LDS buffer 0 (f32 -> bf16 convert in flight)
    #pragma unroll
    for (int t = 0; t < 4; ++t) {
        int chunk = tid + t * 256;
        int r  = chunk >> 3;
        int c4 = (chunk & 7) << 2;
        aReg[t] = *(const float4*)(feat + (size_t)(m0 + r) * C_DIM + c4);
        bReg[t] = *(const float4*)(W    + (size_t)(n0 + r) * C_DIM + c4);
    }
    #pragma unroll
    for (int t = 0; t < 4; ++t) {
        int chunk = tid + t * 256;
        int r  = chunk >> 3;
        int c4 = (chunk & 7) << 2;
        v4bf pa = {(bf16)aReg[t].x, (bf16)aReg[t].y, (bf16)aReg[t].z, (bf16)aReg[t].w};
        v4bf pb = {(bf16)bReg[t].x, (bf16)bReg[t].y, (bf16)bReg[t].z, (bf16)bReg[t].w};
        *(v4bf*)&As[0][r][c4] = pa;
        *(v4bf*)&Bs[0][r][c4] = pb;
    }
    __syncthreads();

    const int KSTEPS = C_DIM / BK;   // 16
    for (int kt = 0; kt < KSTEPS; ++kt) {
        const int cur = kt & 1;
        const int nxt = cur ^ 1;
        if (kt + 1 < KSTEPS) {
            const int kn = (kt + 1) * BK;
            #pragma unroll
            for (int t = 0; t < 4; ++t) {
                int chunk = tid + t * 256;
                int r  = chunk >> 3;
                int c4 = (chunk & 7) << 2;
                aReg[t] = *(const float4*)(feat + (size_t)(m0 + r) * C_DIM + kn + c4);
                bReg[t] = *(const float4*)(W    + (size_t)(n0 + r) * C_DIM + kn + c4);
            }
        }
        // A fragment: row (wave*16+lm), K chunk [lh*16, lh*16+16)
        {
            const bf16* arow = &As[cur][wave * 16 + lm][lh * 16];
            v8bf alo = *(const v8bf*)(arow);
            v8bf ahi = *(const v8bf*)(arow + 8);
            v16bf afrag = __builtin_shufflevector(alo, ahi,
                0,1,2,3,4,5,6,7,8,9,10,11,12,13,14,15);
            #pragma unroll
            for (int t = 0; t < 8; ++t) {
                v16bf bfrag = *(const v16bf*)&Bs[cur][t * 16 + lm][lh * 16];
                acc[t] = __builtin_amdgcn_wmma_f32_16x16x32_bf16(
                    false, afrag, false, bfrag, (short)0, acc[t], false, false);
            }
        }
        if (kt + 1 < KSTEPS) {
            #pragma unroll
            for (int t = 0; t < 4; ++t) {
                int chunk = tid + t * 256;
                int r  = chunk >> 3;
                int c4 = (chunk & 7) << 2;
                v4bf pa = {(bf16)aReg[t].x, (bf16)aReg[t].y, (bf16)aReg[t].z, (bf16)aReg[t].w};
                v4bf pb = {(bf16)bReg[t].x, (bf16)bReg[t].y, (bf16)bReg[t].z, (bf16)bReg[t].w};
                *(v4bf*)&As[nxt][r][c4] = pa;
                *(v4bf*)&Bs[nxt][r][c4] = pb;
            }
        }
        __syncthreads();
    }

    // epilogue: bias + relu + inv_sc scale, transposed bf16 store
    float isc[8];
    #pragma unroll
    for (int r = 0; r < 8; ++r)
        isc[r] = inv_sc[m0 + wave * 16 + lh * 8 + r];
    #pragma unroll
    for (int t = 0; t < 8; ++t) {
        const int c = n0 + t * 16 + lm;
        const float bc = bias[c];
        #pragma unroll
        for (int r = 0; r < 8; ++r) {
            const int j = m0 + wave * 16 + lh * 8 + r;
            float v = acc[t][r] + bc;
            v = v > 0.f ? v : 0.f;
            hs_t[(size_t)c * N_NODES + j] = (bf16)(v * isc[r]);
        }
    }
}

// GEMM-2: out[i][c] = flag ? feat[i][c]
//                          : inv_sr[i] * sum_j bf16(adj[i][j]) * hs_t[c][j] + feat[i][c]
__global__ __launch_bounds__(256)
void gcn_aggregate(const float* __restrict__ adj,
                   const bf16*  __restrict__ hs_t,
                   const float* __restrict__ feat,
                   const float* __restrict__ inv_sr,
                   const int*   __restrict__ flag,
                   float* __restrict__ out) {
    __shared__ __align__(32) bf16 As[2][BM][BK];
    __shared__ __align__(32) bf16 Bs[2][BN][BK];
    const int tid  = threadIdx.x;
    const int wave = tid >> 5;
    const int lane = tid & 31;
    const int lh   = lane >> 4;
    const int lm   = lane & 15;
    const int m0   = blockIdx.y * BM;   // node rows
    const int n0   = blockIdx.x * BN;   // channels

    v8f acc[8] = {};
    float4 aReg[4];

    // B-tile mapping: 8KB per tile; 16B per op; 2 ops per thread
    const int br0 = tid >> 2;                 // chunk0 row 0..63
    const int bo0 = (tid & 3) << 3;           // chunk0 elem offset 0/8/16/24
    const int br1 = (tid + 256) >> 2;         // chunk1 row 64..127
    const int bo1 = bo0;

    // ---- prologue: tile 0 ----
    #pragma unroll
    for (int t = 0; t < 4; ++t) {
        int chunk = tid + t * 256;
        int r  = chunk >> 3;
        int c4 = (chunk & 7) << 2;
        aReg[t] = *(const float4*)(adj + (size_t)(m0 + r) * N_NODES + c4);
    }
#if defined(GCN_ASYNC_LDS)
    gcn_async_b128(hs_t + (size_t)(n0 + br0) * N_NODES + bo0, &Bs[0][br0][bo0]);
    gcn_async_b128(hs_t + (size_t)(n0 + br1) * N_NODES + bo1, &Bs[0][br1][bo1]);
#else
    v16bf hReg;
    const int brow  = tid >> 1;
    const int bhalf = (tid & 1) << 4;
    hReg = *(const v16bf*)(hs_t + (size_t)(n0 + brow) * N_NODES + bhalf);
    *(v16bf*)&Bs[0][brow][bhalf] = hReg;
#endif
    #pragma unroll
    for (int t = 0; t < 4; ++t) {
        int chunk = tid + t * 256;
        int r  = chunk >> 3;
        int c4 = (chunk & 7) << 2;
        v4bf pa = {(bf16)aReg[t].x, (bf16)aReg[t].y, (bf16)aReg[t].z, (bf16)aReg[t].w};
        *(v4bf*)&As[0][r][c4] = pa;
    }
#if defined(GCN_ASYNC_LDS)
    __builtin_amdgcn_s_wait_asynccnt(0);
#endif
    __syncthreads();

    const int KSTEPS = N_NODES / BK;   // 256
    for (int kt = 0; kt < KSTEPS; ++kt) {
        const int cur = kt & 1;
        const int nxt = cur ^ 1;
        if (kt + 1 < KSTEPS) {
            const int kn = (kt + 1) * BK;
#if defined(GCN_ASYNC_LDS)
            // async DMA next B tile straight into LDS (overlaps the WMMAs below)
            gcn_async_b128(hs_t + (size_t)(n0 + br0) * N_NODES + kn + bo0,
                           &Bs[nxt][br0][bo0]);
            gcn_async_b128(hs_t + (size_t)(n0 + br1) * N_NODES + kn + bo1,
                           &Bs[nxt][br1][bo1]);
#else
            hReg = *(const v16bf*)(hs_t + (size_t)(n0 + brow) * N_NODES + kn + bhalf);
#endif
            #pragma unroll
            for (int t = 0; t < 4; ++t) {
                int chunk = tid + t * 256;
                int r  = chunk >> 3;
                int c4 = (chunk & 7) << 2;
                aReg[t] = *(const float4*)(adj + (size_t)(m0 + r) * N_NODES + kn + c4);
            }
        }
        {
            const bf16* arow = &As[cur][wave * 16 + lm][lh * 16];
            v8bf alo = *(const v8bf*)(arow);
            v8bf ahi = *(const v8bf*)(arow + 8);
            v16bf afrag = __builtin_shufflevector(alo, ahi,
                0,1,2,3,4,5,6,7,8,9,10,11,12,13,14,15);
            #pragma unroll
            for (int t = 0; t < 8; ++t) {
                v16bf bfrag = *(const v16bf*)&Bs[cur][t * 16 + lm][lh * 16];
                acc[t] = __builtin_amdgcn_wmma_f32_16x16x32_bf16(
                    false, afrag, false, bfrag, (short)0, acc[t], false, false);
            }
        }
        if (kt + 1 < KSTEPS) {
            #pragma unroll
            for (int t = 0; t < 4; ++t) {
                int chunk = tid + t * 256;
                int r  = chunk >> 3;
                int c4 = (chunk & 7) << 2;
                v4bf pa = {(bf16)aReg[t].x, (bf16)aReg[t].y, (bf16)aReg[t].z, (bf16)aReg[t].w};
                *(v4bf*)&As[nxt][r][c4] = pa;
            }
#if !defined(GCN_ASYNC_LDS)
            *(v16bf*)&Bs[nxt][brow][bhalf] = hReg;
#endif
        }
#if defined(GCN_ASYNC_LDS)
        __builtin_amdgcn_s_wait_asynccnt(0);
#endif
        __syncthreads();
    }

    // epilogue: row scale + identity add + has_zero select
    const int fz = *flag;
    float isr[8];
    #pragma unroll
    for (int r = 0; r < 8; ++r)
        isr[r] = inv_sr[m0 + wave * 16 + lh * 8 + r];
    #pragma unroll
    for (int t = 0; t < 8; ++t) {
        const int c = n0 + t * 16 + lm;
        #pragma unroll
        for (int r = 0; r < 8; ++r) {
            const int i = m0 + wave * 16 + lh * 8 + r;
            const size_t idx = (size_t)i * C_DIM + c;
            const float idv = feat[idx];
            out[idx] = fz ? idv : fmaf(acc[t][r], isr[r], idv);
        }
    }
}

extern "C" void kernel_launch(void* const* d_in, const int* in_sizes, int n_in,
                              void* d_out, int out_size, void* d_ws, size_t ws_size,
                              hipStream_t stream) {
    const float* feat = (const float*)d_in[0];   // [8192,512]
    const float* adj  = (const float*)d_in[1];   // [8192,8192]
    const float* W    = (const float*)d_in[2];   // [512,512]
    const float* bias = (const float*)d_in[3];   // [512]
    float* out = (float*)d_out;

    char* ws = (char*)d_ws;
    int*   flag   = (int*)ws;
    float* inv_sr = (float*)(ws + WS_OFF_INVSR);
    float* inv_sc = (float*)(ws + WS_OFF_INVSC);
    bf16*  hs_t   = (bf16*)(ws + WS_OFF_HST);

    gcn_init_flag<<<1, 1, 0, stream>>>(flag);
    gcn_row_degree<<<N_NODES, 256, 0, stream>>>(adj, inv_sr, flag);
    gcn_col_degree<<<N_NODES / 256, 256, 0, stream>>>(adj, inv_sc, flag);
    gcn_feat_xform<<<dim3(C_DIM / BN, N_NODES / BM), 256, 0, stream>>>(
        feat, W, bias, inv_sc, hs_t);
    gcn_aggregate<<<dim3(C_DIM / BN, N_NODES / BM), 256, 0, stream>>>(
        adj, hs_t, feat, inv_sr, flag, out);
}
